// MarketLSTM_51101520888117
// MI455X (gfx1250) — matrix operand order repaired and checked by reading, not compiled, not run on hardware
//
#include <hip/hip_runtime.h>
#include <hip/hip_bf16.h>

// ---------------- problem constants ----------------
#define BATCH   1024
#define SEQ     512
#define INPUT   32
#define HIDDEN  128
#define G4      (4*HIDDEN)   // 512 gate rows
#define FC1N    64

typedef __attribute__((ext_vector_type(16))) _Float16 v16h;
typedef __attribute__((ext_vector_type(8)))  _Float16 v8h;
typedef __attribute__((ext_vector_type(8)))  float    v8f;

// ---------------- weight packing -------------------
// Packs row-major W[R][K] (fp32) into WMMA 16-bit A-matrix fragments (f16):
// frag(m, kt) = 32 lanes x 16 halves, lane-contiguous 32B.
// A 16x32 layout (contiguous K order): lanes 0-15 row M=lane, V0..V7 K=0..15;
// lanes 16-31 row M=lane-16, V0..V7 K=16..31.
__global__ void pack_w_kernel(const float* __restrict__ src,
                              _Float16*    __restrict__ dst,
                              int R, int K) {
  int tid = blockIdx.x * blockDim.x + threadIdx.x;
  int total = R * K;
  if (tid >= total) return;
  int e    = tid & 15;
  int lane = (tid >> 4) & 31;
  int fk   = tid >> 9;           // m*nkt + kt
  int nkt  = K >> 5;
  int kt   = fk % nkt;
  int m    = fk / nkt;
  bool lo  = lane < 16;
  int row  = m * 16 + (lane & 15);
  int off  = lo ? e : e + 16;
  int col  = kt * 32 + off;
  dst[tid] = (_Float16)src[row * K + col];
}

__global__ void bias_kernel(const float* bi0, const float* bh0,
                            const float* bi1, const float* bh1,
                            float* b0, float* b1) {
  int j = blockIdx.x * blockDim.x + threadIdx.x;
  if (j < G4)            b0[j]       = bi0[j]       + bh0[j];
  else if (j < 2 * G4)   b1[j - G4]  = bi1[j - G4]  + bh1[j - G4];
}

// ---------------- helpers --------------------------
// Plain load: used for intentionally loop-hoisted (register-resident) fragments.
__device__ __forceinline__ v16h ld_frag16(const _Float16* p) {
  union { v16h v; v8h h[2]; } u;
  u.h[0] = *(const v8h*)p;
  u.h[1] = *(const v8h*)(p + 8);
  return u.v;
}

// Volatile load: streamed every iteration from L2 (LLVM cannot hoist/spill-cache it).
__device__ __forceinline__ v16h ld_frag16_stream(const _Float16* p) {
  union { v16h v; v8h h[2]; } u;
  u.h[0] = *(const volatile v8h*)p;
  u.h[1] = *(const volatile v8h*)(p + 8);
  return u.v;
}

// Branch-free activations on native TRANS ops (v_exp_f32 / v_rcp_f32).
// sigmoid(x) = 1/(1+e^-x); tanh(x) = 2*sigmoid(2x)-1. Saturation at +/-inf is exact.
#define LOG2E 1.4426950408889634f
__device__ __forceinline__ float fast_sig(float x) {
  float e = __builtin_amdgcn_exp2f(-LOG2E * x);
  return __builtin_amdgcn_rcpf(1.f + e);
}
__device__ __forceinline__ float fast_tanh(float x) {
  float e = __builtin_amdgcn_exp2f(-2.f * LOG2E * x);
  return fmaf(2.f, __builtin_amdgcn_rcpf(1.f + e), -1.f);
}

#define LROW 136   // padded LDS row stride in f16 (272B, 16B aligned per fragment)

// ---------------- fused persistent LSTM ------------
// grid = 64 blocks (16 batch rows each), block = 256 threads = 8 waves (wave32).
// Wave w owns hidden slice [w*16, w*16+16): gate tiles {w, w+8, w+16, w+24} (i,f,g,o).
// Layer-0 weights (20 frags = 160 VGPRs) are register-resident; layer-1 weights
// (32 frags) are streamed from the 192MB L2 each step via volatile loads.
__global__ __launch_bounds__(256)
void lstm_fused_kernel(const float* __restrict__ x,
                       const _Float16* __restrict__ wih0,
                       const _Float16* __restrict__ whh0,
                       const _Float16* __restrict__ wih1,
                       const _Float16* __restrict__ whh1,
                       const float* __restrict__ bias0,
                       const float* __restrict__ bias1,
                       const float* __restrict__ fc1_w,
                       const float* __restrict__ fc1_b,
                       const float* __restrict__ fc2_w,
                       const float* __restrict__ fc2_b,
                       float* __restrict__ out) {
  __shared__ __align__(16) _Float16 h0_sh[16 * LROW];
  __shared__ __align__(16) _Float16 h1_sh[16 * LROW];

  const int tid  = threadIdx.x;
  const int w    = tid >> 5;          // wave id 0..7 (hidden slice)
  const int lane = tid & 31;
  const int n    = lane & 15;         // batch column within tile
  const bool lo  = lane < 16;
  const int b0   = blockIdx.x * 16;
  const int b    = b0 + n;
  const int hoff = lo ? 0 : 16;       // B-fragment K offset within 32-K tile
  const int kbase = w * 16;           // hidden base of this wave

  // zero LDS h state
  for (int i = tid; i < 16 * LROW; i += 256) { h0_sh[i] = (_Float16)0.f; h1_sh[i] = (_Float16)0.f; }
  __syncthreads();

  // ---- layer-0 weights: register resident (loaded once) ----
  v16h w0x[4];        // w_ih0 frag per gate (K=32 -> 1 ktile)
  v16h w0h[4][4];     // w_hh0 frags [gate][kt]
#pragma unroll
  for (int g = 0; g < 4; ++g) {
    const int m = g * 8 + w;
    w0x[g] = ld_frag16(wih0 + ((size_t)m * 32 + lane) * 16);
#pragma unroll
    for (int kt = 0; kt < 4; ++kt)
      w0h[g][kt] = ld_frag16(whh0 + ((size_t)(m * 4 + kt) * 32 + lane) * 16);
  }

  // bias registers: biasX[g][r] seeds the f32 accumulator (C matrix)
  float bj0[4][8], bj1[4][8];
#pragma unroll
  for (int g = 0; g < 4; ++g)
#pragma unroll
    for (int r = 0; r < 8; ++r) {
      int row = g * HIDDEN + kbase + (lo ? r : 8 + r);
      bj0[g][r] = bias0[row];
      bj1[g][r] = bias1[row];
    }

  float c0s[8], c1s[8];
#pragma unroll
  for (int r = 0; r < 8; ++r) { c0s[r] = 0.f; c1s[r] = 0.f; }

  const float* xbase = x + (size_t)b * SEQ * INPUT + (lo ? 0 : 16);

  for (int t = 0; t < SEQ; ++t) {
    // ---- x_t B-fragment (16 contiguous f32 -> f16) ----
    const float* xp = xbase + (size_t)t * INPUT;
    if (t + 1 < SEQ) __builtin_prefetch(xbase + (size_t)(t + 1) * INPUT, 0, 1);
    union { v16h v; _Float16 s[16]; } xb;
    {
      const float4* f4 = (const float4*)xp;
#pragma unroll
      for (int q = 0; q < 4; ++q) {
        float4 fv = f4[q];
        xb.s[4 * q + 0] = (_Float16)fv.x; xb.s[4 * q + 1] = (_Float16)fv.y;
        xb.s[4 * q + 2] = (_Float16)fv.z; xb.s[4 * q + 3] = (_Float16)fv.w;
      }
    }

    // ---- h0_{t-1} B-fragments from LDS ----
    v16h hA[4];
#pragma unroll
    for (int kt = 0; kt < 4; ++kt)
      hA[kt] = ld_frag16(&h0_sh[n * LROW + kt * 32 + hoff]);

    // ---- layer 0 gates:  W_ih0 * x_t  +  W_hh0 * h0_{t-1}  + bias ----
    v8f acc[4];
#pragma unroll
    for (int g = 0; g < 4; ++g) {
#pragma unroll
      for (int r = 0; r < 8; ++r) acc[g][r] = bj0[g][r];
      acc[g] = __builtin_amdgcn_wmma_f32_16x16x32_f16(false, w0x[g], false, xb.v,
                                                      (short)0, acc[g], false, false);
#pragma unroll
      for (int kt = 0; kt < 4; ++kt)
        acc[g] = __builtin_amdgcn_wmma_f32_16x16x32_f16(false, w0h[g][kt], false, hA[kt],
                                                        (short)0, acc[g], false, false);
    }

    // ---- layer 0 cell (branch-free TRANS activations) ----
    float hn0[8];
#pragma unroll
    for (int r = 0; r < 8; ++r) {
      float iv = fast_sig(acc[0][r]);
      float fv = fast_sig(acc[1][r]);
      float gv = fast_tanh(acc[2][r]);
      float ov = fast_sig(acc[3][r]);
      c0s[r] = fv * c0s[r] + iv * gv;
      hn0[r] = ov * fast_tanh(c0s[r]);
    }
    __syncthreads();                      // all reads of h0_{t-1} done
#pragma unroll
    for (int r = 0; r < 8; ++r)
      h0_sh[n * LROW + kbase + (lo ? r : 8 + r)] = (_Float16)hn0[r];
    __syncthreads();                      // h0_t visible

    // ---- layer 1 B-fragments: h0_t and h1_{t-1} ----
    v16h hB[4];
#pragma unroll
    for (int kt = 0; kt < 4; ++kt) {
      hA[kt] = ld_frag16(&h0_sh[n * LROW + kt * 32 + hoff]);   // h0_t
      hB[kt] = ld_frag16(&h1_sh[n * LROW + kt * 32 + hoff]);   // h1_{t-1}
    }

    // ---- layer 1 gates:  W_ih1 * h0_t + W_hh1 * h1_{t-1} + bias ----
    // Weight fragments streamed from L2 each step (volatile: not hoistable).
#pragma unroll
    for (int g = 0; g < 4; ++g) {
#pragma unroll
      for (int r = 0; r < 8; ++r) acc[g][r] = bj1[g][r];
      const int m = g * 8 + w;
#pragma unroll
      for (int kt = 0; kt < 4; ++kt) {
        v16h ai = ld_frag16_stream(wih1 + ((size_t)(m * 4 + kt) * 32 + lane) * 16);
        acc[g] = __builtin_amdgcn_wmma_f32_16x16x32_f16(false, ai, false, hA[kt],
                                                        (short)0, acc[g], false, false);
        v16h ah = ld_frag16_stream(whh1 + ((size_t)(m * 4 + kt) * 32 + lane) * 16);
        acc[g] = __builtin_amdgcn_wmma_f32_16x16x32_f16(false, ah, false, hB[kt],
                                                        (short)0, acc[g], false, false);
      }
    }

    // ---- layer 1 cell ----
    float hn1[8];
#pragma unroll
    for (int r = 0; r < 8; ++r) {
      float iv = fast_sig(acc[0][r]);
      float fv = fast_sig(acc[1][r]);
      float gv = fast_tanh(acc[2][r]);
      float ov = fast_sig(acc[3][r]);
      c1s[r] = fv * c1s[r] + iv * gv;
      hn1[r] = ov * fast_tanh(c1s[r]);
    }
    __syncthreads();                      // all reads of h1_{t-1} done
#pragma unroll
    for (int r = 0; r < 8; ++r)
      h1_sh[n * LROW + kbase + (lo ? r : 8 + r)] = (_Float16)hn1[r];
    __syncthreads();                      // h1_t visible
  }

  // ---- FC head from final h1 (8 MFLOP total; negligible) ----
  if (tid < 16) {
    int nb = tid;
    float o = fc2_b[0];
    for (int f = 0; f < FC1N; ++f) {
      float s = fc1_b[f];
      for (int k = 0; k < HIDDEN; ++k)
        s += (float)h1_sh[nb * LROW + k] * fc1_w[f * HIDDEN + k];
      s = fmaxf(s, 0.f);
      o += fc2_w[f] * s;
    }
    out[b0 + nb] = o;
  }
}

// ---------------- launcher -------------------------
extern "C" void kernel_launch(void* const* d_in, const int* in_sizes, int n_in,
                              void* d_out, int out_size, void* d_ws, size_t ws_size,
                              hipStream_t stream) {
  const float* x     = (const float*)d_in[0];
  const float* w_ih0 = (const float*)d_in[1];
  const float* w_hh0 = (const float*)d_in[2];
  const float* b_ih0 = (const float*)d_in[3];
  const float* b_hh0 = (const float*)d_in[4];
  const float* w_ih1 = (const float*)d_in[5];
  const float* w_hh1 = (const float*)d_in[6];
  const float* b_ih1 = (const float*)d_in[7];
  const float* b_hh1 = (const float*)d_in[8];
  const float* fc1_w = (const float*)d_in[9];
  const float* fc1_b = (const float*)d_in[10];
  const float* fc2_w = (const float*)d_in[11];
  const float* fc2_b = (const float*)d_in[12];
  float* out = (float*)d_out;

  // workspace layout (f16 fragment-packed weights + combined biases)
  _Float16* ws16   = (_Float16*)d_ws;
  _Float16* p_wih0 = ws16;                 // 512*32   = 16384 halves
  _Float16* p_whh0 = p_wih0 + 16384;       // 512*128  = 65536
  _Float16* p_wih1 = p_whh0 + 65536;
  _Float16* p_whh1 = p_wih1 + 65536;
  float*    p_b0   = (float*)((char*)d_ws + 425984);   // 256B aligned
  float*    p_b1   = p_b0 + G4;

  pack_w_kernel<<<(G4 * INPUT  + 255) / 256, 256, 0, stream>>>(w_ih0, p_wih0, G4, INPUT);
  pack_w_kernel<<<(G4 * HIDDEN + 255) / 256, 256, 0, stream>>>(w_hh0, p_whh0, G4, HIDDEN);
  pack_w_kernel<<<(G4 * HIDDEN + 255) / 256, 256, 0, stream>>>(w_ih1, p_wih1, G4, HIDDEN);
  pack_w_kernel<<<(G4 * HIDDEN + 255) / 256, 256, 0, stream>>>(w_hh1, p_whh1, G4, HIDDEN);
  bias_kernel  <<<(2 * G4 + 255) / 256, 256, 0, stream>>>(b_ih0, b_hh0, b_ih1, b_hh1, p_b0, p_b1);

  lstm_fused_kernel<<<BATCH / 16, 256, 0, stream>>>(
      x, p_wih0, p_whh0, p_wih1, p_whh1, p_b0, p_b1,
      fc1_w, fc1_b, fc2_w, fc2_b, out);

  (void)in_sizes; (void)n_in; (void)out_size; (void)ws_size;
}
